// Pseudo3DConv_16166256902407
// MI455X (gfx1250) — compile-verified, matrix operand order checked by reading, not run on hardware
//
#include <hip/hip_runtime.h>
#include <hip/hip_bf16.h>
#include <math.h>

#define N_PTS 8192
#define KNN   32

typedef __attribute__((ext_vector_type(16))) _Float16 v16h;
typedef __attribute__((ext_vector_type(8)))  _Float16 v8h;
typedef __attribute__((ext_vector_type(8)))  float    v8f;

// CDNA5 WMMA: f16 A (16x32) x f16 B (32x16) -> f32 C/D (16x16)
__device__ __forceinline__ v8f wmma_f16(v16h a, v16h b, v8f c) {
    return __builtin_amdgcn_wmma_f32_16x16x32_f16(
        /*neg_a=*/false, a, /*neg_b=*/false, b,
        /*c_mod=*/(short)0, c, /*reuse_a=*/false, /*reuse_b=*/false);
}

// Per ISA 7.12.2, a 16-bit A/B fragment's K indices for a lane in half `h`
// are two contiguous 8-element chunks: {h*8 .. h*8+7} and {16+h*8 .. 16+h*8+7}.
// Build a fragment from two 16-byte (v8h) loads.
__device__ __forceinline__ v16h frag_from_chunks(const _Float16* base) {
    v8h c0 = *(const v8h*)(base);        // k = off .. off+7
    v8h c1 = *(const v8h*)(base + 16);   // k = off+16 .. off+23
    v16h f;
    #pragma unroll
    for (int e = 0; e < 8; ++e) { f[e] = c0[e]; f[8 + e] = c1[e]; }
    return f;
}

// ---------------------------------------------------------------------------
// Kernel 0a: transpose + f16-convert image features: img[32,N] -> featsT[N,32].
// ---------------------------------------------------------------------------
__global__ __launch_bounds__(256) void feat_transpose_kernel(
    const float* __restrict__ img,      // [32, N]
    _Float16*    __restrict__ featsT)   // [N, 32]
{
    const int n = blockIdx.x * 256 + threadIdx.x;
    _Float16 row[32];
    #pragma unroll
    for (int c = 0; c < 32; ++c)                 // coalesced reads per c
        row[c] = (_Float16)img[c * N_PTS + n];
    v8h* dst = (v8h*)(featsT + (size_t)n * 32);  // 64B/thread vector stores
    #pragma unroll
    for (int q = 0; q < 4; ++q) {
        v8h v;
        #pragma unroll
        for (int e = 0; e < 8; ++e) v[e] = row[q * 8 + e];
        dst[q] = v;
    }
}

// ---------------------------------------------------------------------------
// Kernel 0b: one-shot f16 conversion of the WMMA weight matrices.
// ---------------------------------------------------------------------------
__global__ __launch_bounds__(256) void weight_cvt_kernel(
    const float* __restrict__ W1, const float* __restrict__ W2,
    const float* __restrict__ Wf,
    _Float16* __restrict__ W1h, _Float16* __restrict__ W2h,
    _Float16* __restrict__ Wfh)
{
    const int t = blockIdx.x * 256 + threadIdx.x;   // grid covers 32768
    if (t < 64 * 32)   W1h[t] = (_Float16)W1[t];
    if (t < 128 * 64)  W2h[t] = (_Float16)W2[t];
    if (t < 128 * 256) Wfh[t] = (_Float16)Wf[t];
}

// ---------------------------------------------------------------------------
// Kernel 1: per-point KNN (top-32 of 8192) + softmax weights.
// Distances in LDS; lazy-rescan selection: 2 barriers/round, wave32 shuffles.
// ---------------------------------------------------------------------------
__global__ __launch_bounds__(256) void knn_kernel(
    const float* __restrict__ cloud,  // [N,3]
    int*   __restrict__ idx_out,      // [N,32]
    float* __restrict__ w_out)        // [N,32]
{
    __shared__ float d2s[N_PTS];      // 32KB of the 320KB WGP LDS
    __shared__ float wmin[8];
    __shared__ int   widx[8];
    __shared__ float sel_d[KNN];
    __shared__ int   sel_i[KNN];
    __shared__ int   bcast_i;

    const int n    = blockIdx.x;
    const int tid  = threadIdx.x;
    const int lane = tid & 31;
    const int wid  = tid >> 5;

    const float px = cloud[n * 3 + 0];
    const float py = cloud[n * 3 + 1];
    const float pz = cloud[n * 3 + 2];

    for (int j = tid; j < N_PTS; j += 256) {
        if (j + 2048 < N_PTS)
            __builtin_prefetch(&cloud[(j + 2048) * 3], 0, 0);  // global_prefetch_b8
        float dx = cloud[j * 3 + 0] - px;
        float dy = cloud[j * 3 + 1] - py;
        float dz = cloud[j * 3 + 2] - pz;
        d2s[j] = dx * dx + dy * dy + dz * dz;
    }
    __syncthreads();

    float bm = 3.0e38f; int bi = tid;
    for (int j = tid; j < N_PTS; j += 256) {
        float v = d2s[j];
        if (v < bm) { bm = v; bi = j; }
    }

    for (int r = 0; r < KNN; ++r) {
        float v = bm; int i = bi;
        #pragma unroll
        for (int off = 16; off > 0; off >>= 1) {
            float ov = __shfl_xor(v, off, 32);
            int   oi = __shfl_xor(i, off, 32);
            if (ov < v) { v = ov; i = oi; }
        }
        if (lane == 0) { wmin[wid] = v; widx[wid] = i; }
        __syncthreads();
        if (tid == 0) {
            float gv = wmin[0]; int gi = widx[0];
            #pragma unroll
            for (int w = 1; w < 8; ++w)
                if (wmin[w] < gv) { gv = wmin[w]; gi = widx[w]; }
            sel_i[r] = gi;
            sel_d[r] = sqrtf(fmaxf(gv, 0.0f));
            d2s[gi]  = 3.0e38f;     // mask winner
            bcast_i  = gi;
        }
        __syncthreads();
        if ((bcast_i & 255) == tid) {   // only the winner's owner rescans
            bm = 3.0e38f; bi = tid;
            for (int j = tid; j < N_PTS; j += 256) {
                float v2 = d2s[j];
                if (v2 < bm) { bm = v2; bi = j; }
            }
        }
    }

    if (tid < KNN) {
        float e = expf(sel_d[0] - sel_d[tid]);
        float s = e;
        #pragma unroll
        for (int off = 16; off > 0; off >>= 1)
            s += __shfl_xor(s, off, 32);
        idx_out[n * KNN + tid] = sel_i[tid];
        w_out[n * KNN + tid]   = e / s;
    }
}

// ---------------------------------------------------------------------------
// Kernel 2: neighbor MLP 32 -> 64 -> 128 with WMMA, weighted max over K.
// One wave per point; all A/B fragments built from b128 loads of f16 data.
// ---------------------------------------------------------------------------
__global__ __launch_bounds__(256) void mlp_sur_kernel(
    const _Float16* __restrict__ featsT, // [N, 32] f16
    const int*      __restrict__ idx,    // [N, 32]
    const float*    __restrict__ wgt,    // [N, 32]
    const _Float16* __restrict__ W1h,    // [64, 32] f16
    const float*    __restrict__ B1,     // [64]
    const _Float16* __restrict__ W2h,    // [128, 64] f16
    const float*    __restrict__ B2,     // [128]
    _Float16*       __restrict__ finalbuf) // [N, 256] f16, cols 0..127
{
    __shared__ int      idx_s[8][KNN];
    __shared__ float    w_s[8][KNN];
    __shared__ _Float16 h1_s[8][32][64];

    const int wave    = threadIdx.x >> 5;
    const int lane    = threadIdx.x & 31;
    const int n       = blockIdx.x * 8 + wave;
    const int halfsel = lane >> 4;
    const int l15     = lane & 15;

    idx_s[wave][lane] = idx[n * KNN + lane];
    w_s[wave][lane]   = wgt[n * KNN + lane];
    __syncthreads();

    // ---- A1: 2 tiles of [16 neighbors x 32 channels]; 2x b128 gather each
    v16h a1[2];
    #pragma unroll
    for (int t = 0; t < 2; ++t) {
        const int nbr = idx_s[wave][t * 16 + l15];
        a1[t] = frag_from_chunks(featsT + (size_t)nbr * 32 + halfsel * 8);
    }
    // ---- B1: W1^T [32 x 64] -> 4 N-tiles, b128 loads
    v16h b1f[4];
    #pragma unroll
    for (int u = 0; u < 4; ++u)
        b1f[u] = frag_from_chunks(W1h + (u * 16 + l15) * 32 + halfsel * 8);

    // ---- layer 1 (8x v_wmma_f32_16x16x32_f16) + bias
    v8f c1acc[2][4];
    #pragma unroll
    for (int t = 0; t < 2; ++t)
        #pragma unroll
        for (int u = 0; u < 4; ++u) {
            v8f acc = {};
            acc = wmma_f16(a1[t], b1f[u], acc);
            const float bo = B1[u * 16 + l15];
            #pragma unroll
            for (int r = 0; r < 8; ++r) acc[r] += bo;
            c1acc[t][u] = acc;
        }

    // ---- stage h1 to LDS (C-layout -> row-major f16), reload in A-layout
    #pragma unroll
    for (int t = 0; t < 2; ++t)
        #pragma unroll
        for (int u = 0; u < 4; ++u)
            #pragma unroll
            for (int r = 0; r < 8; ++r) {
                const int m = t * 16 + r + halfsel * 8;
                h1_s[wave][m][u * 16 + l15] = (_Float16)c1acc[t][u][r];
            }
    __syncthreads();

    v16h a2[2][2];   // [M-tile][K-chunk]; ds_load_b128 pairs
    #pragma unroll
    for (int t = 0; t < 2; ++t)
        #pragma unroll
        for (int kc = 0; kc < 2; ++kc)
            a2[t][kc] = frag_from_chunks(
                &h1_s[wave][t * 16 + l15][kc * 32 + halfsel * 8]);

    // ---- layer 2 (32 WMMAs) + distance-weighted max over 32 neighbor rows
    #pragma unroll
    for (int v = 0; v < 8; ++v) {
        const int o = v * 16 + l15;
        v16h b2f[2];
        #pragma unroll
        for (int kc = 0; kc < 2; ++kc)
            b2f[kc] = frag_from_chunks(W2h + o * 64 + kc * 32 + halfsel * 8);
        const float bo = B2[o];
        float pm = -3.0e38f;
        #pragma unroll
        for (int t = 0; t < 2; ++t) {
            v8f acc = {};
            acc = wmma_f16(a2[t][0], b2f[0], acc);
            acc = wmma_f16(a2[t][1], b2f[1], acc);
            #pragma unroll
            for (int r = 0; r < 8; ++r) {
                const int m = t * 16 + r + halfsel * 8;       // neighbor index
                pm = fmaxf(pm, (acc[r] + bo) * w_s[wave][m]);
            }
        }
        pm = fmaxf(pm, __shfl_xor(pm, 16, 32));  // merge m-halves (same column)
        if (halfsel == 0)
            finalbuf[n * 256 + v * 16 + l15] = (_Float16)pm;
    }
}

// ---------------------------------------------------------------------------
// Kernel 3: point branch 3 -> 64 -> 128 (tiny, plain VALU), f16 output.
// ---------------------------------------------------------------------------
__global__ __launch_bounds__(256) void point_branch_kernel(
    const float* __restrict__ cloud,
    const float* __restrict__ Wp1, const float* __restrict__ bp1,
    const float* __restrict__ Wp2, const float* __restrict__ bp2,
    _Float16* __restrict__ finalbuf)
{
    const int n = blockIdx.x * blockDim.x + threadIdx.x;
    if (n >= N_PTS) return;
    const float px = cloud[n * 3 + 0];
    const float py = cloud[n * 3 + 1];
    const float pz = cloud[n * 3 + 2];
    float h[64];
    #pragma unroll
    for (int o = 0; o < 64; ++o)
        h[o] = Wp1[o * 3 + 0] * px + Wp1[o * 3 + 1] * py + Wp1[o * 3 + 2] * pz + bp1[o];
    for (int p = 0; p < 128; ++p) {
        float s = bp2[p];
        #pragma unroll 8
        for (int o = 0; o < 64; ++o) s += Wp2[p * 64 + o] * h[o];
        finalbuf[n * 256 + 128 + p] = (_Float16)s;
    }
}

// ---------------------------------------------------------------------------
// Kernel 4: final GEMM [8192,256] x [256,128] + bias, transposed f32 store.
// All fragments are direct b128 f16 loads; zero cvt in the loop.
// ---------------------------------------------------------------------------
__global__ __launch_bounds__(256) void final_gemm_kernel(
    const _Float16* __restrict__ finalbuf,  // [N, 256] f16
    const _Float16* __restrict__ Wfh,       // [128, 256] f16
    const float*    __restrict__ bf,        // [128]
    float* __restrict__ out)                // [128, N]
{
    const int wave    = threadIdx.x >> 5;
    const int lane    = threadIdx.x & 31;
    const int halfsel = lane >> 4;
    const int l15     = lane & 15;
    const int mbase   = blockIdx.x * 16;
    const int o       = wave * 16 + l15;
    const int row     = mbase + l15;

    v8f acc = {};
    #pragma unroll
    for (int kc = 0; kc < 8; ++kc) {
        v16h a = frag_from_chunks(finalbuf + (size_t)row * 256 + kc * 32 + halfsel * 8);
        v16h b = frag_from_chunks(Wfh      + (size_t)o   * 256 + kc * 32 + halfsel * 8);
        acc = wmma_f16(a, b, acc);
    }
    const float bo = bf[o];
    #pragma unroll
    for (int r = 0; r < 8; ++r) {
        const int m = mbase + r + halfsel * 8;
        out[o * N_PTS + m] = acc[r] + bo;
    }
}

// ---------------------------------------------------------------------------
extern "C" void kernel_launch(void* const* d_in, const int* in_sizes, int n_in,
                              void* d_out, int out_size, void* d_ws, size_t ws_size,
                              hipStream_t stream) {
    (void)in_sizes; (void)n_in; (void)out_size; (void)ws_size;

    const float* img   = (const float*)d_in[0];
    const float* cloud = (const float*)d_in[1];
    const float* W1    = (const float*)d_in[2];
    const float* B1    = (const float*)d_in[3];
    const float* W2    = (const float*)d_in[4];
    const float* B2    = (const float*)d_in[5];
    const float* Wp1   = (const float*)d_in[6];
    const float* bp1   = (const float*)d_in[7];
    const float* Wp2   = (const float*)d_in[8];
    const float* bp2   = (const float*)d_in[9];
    const float* Wf    = (const float*)d_in[10];
    const float* bf    = (const float*)d_in[11];
    float* out = (float*)d_out;

    // Workspace (all offsets 64B-aligned):
    //  idx [N,32] i32 (1MB) | w [N,32] f32 (1MB) | final [N,256] f16 (4MB)
    //  featsT [N,32] f16 (512KB) | W1h (4KB) | W2h (16KB) | Wfh (64KB)
    char* ws = (char*)d_ws;
    size_t off = 0;
    int*      idx_ws = (int*)(ws + off);       off += (size_t)N_PTS * KNN * 4;
    float*    w_ws   = (float*)(ws + off);     off += (size_t)N_PTS * KNN * 4;
    _Float16* fin_ws = (_Float16*)(ws + off);  off += (size_t)N_PTS * 256 * 2;
    _Float16* ft_ws  = (_Float16*)(ws + off);  off += (size_t)N_PTS * 32 * 2;
    _Float16* W1h    = (_Float16*)(ws + off);  off += (size_t)64 * 32 * 2;
    _Float16* W2h    = (_Float16*)(ws + off);  off += (size_t)128 * 64 * 2;
    _Float16* Wfh    = (_Float16*)(ws + off);  off += (size_t)128 * 256 * 2;

    feat_transpose_kernel<<<N_PTS / 256, 256, 0, stream>>>(img, ft_ws);
    weight_cvt_kernel<<<128, 256, 0, stream>>>(W1, W2, Wf, W1h, W2h, Wfh);
    knn_kernel<<<N_PTS, 256, 0, stream>>>(cloud, idx_ws, w_ws);
    mlp_sur_kernel<<<N_PTS / 8, 256, 0, stream>>>(ft_ws, idx_ws, w_ws,
                                                  W1h, B1, W2h, B2, fin_ws);
    point_branch_kernel<<<N_PTS / 256, 256, 0, stream>>>(cloud, Wp1, bp1, Wp2, bp2, fin_ws);
    final_gemm_kernel<<<N_PTS / 16, 256, 0, stream>>>(fin_ws, Wfh, bf, out);
}
